// Encoder_29867202576652
// MI455X (gfx1250) — compile-verified, matrix operand order checked by reading, not run on hardware
//
#include <hip/hip_runtime.h>
#include <hip/hip_bf16.h>
#include <stdint.h>

// ---------------------------------------------------------------------------
// Transformer encoder for gfx1250 (MI455X):
//  - all matmuls via v_wmma_f32_16x16x32_f16
//  - BOTH GEMM tiles staged with global_load_async_to_lds_b128 (ASYNCcnt path)
//  - weights pre-converted once to f16, pre-transposed [N][K]
//  - f16 shadow activations maintained by every producer
// ---------------------------------------------------------------------------

typedef __attribute__((ext_vector_type(16))) _Float16 v16h;
typedef __attribute__((ext_vector_type(8)))  float    v8f;
typedef __attribute__((ext_vector_type(4)))  float    f32x4;

#define D_DIM 1024
#define S_DIM 1024
#define F_DIM 4096
#define B_DIM 8
#define H_DIM 16
#define L_DIM 6
#define NT    (B_DIM * S_DIM)   // 8192 token rows

union F16x16 { v16h v; unsigned int u[8]; };
union Pack2  { unsigned int u; _Float16 h[2]; };

__device__ __forceinline__ unsigned int pack2(float a, float b) {
    Pack2 p; p.h[0] = (_Float16)a; p.h[1] = (_Float16)b; return p.u;
}

// K-index of element pair held in VGPR v for the f16 A/B fragment layout
// (CDNA5 ISA 7.12.2: 16-bit A-matrix 16x32).
__device__ __forceinline__ int kmap(int v, int half) {
    return (v < 4) ? (half * 8 + 2 * v) : (16 + half * 8 + 2 * (v - 4));
}

// ---------------------------------------------------------------------------
// Weight convert + transpose: Wt[n][k] (f16) = W[k][n] (fp32); one 32x32 tile
// per block, LDS staged so both global streams are coalesced. grid.z = layer.
// ---------------------------------------------------------------------------
__global__ __launch_bounds__(256) void wconv_kernel(const float* __restrict__ W,
                                                    _Float16* __restrict__ Wt,
                                                    int K, int N) {
    __shared__ float tile[32][33];
    const int tid = threadIdx.x;
    const int k0  = blockIdx.x * 32;
    const int n0  = blockIdx.y * 32;
    const size_t lofs = (size_t)blockIdx.z * K * N;
    const int r  = tid >> 3;         // 0..31
    const int c4 = (tid & 7) * 4;    // 0,4,..,28

    f32x4 vv = *(const f32x4*)&W[lofs + (size_t)(k0 + r) * N + n0 + c4];
    tile[r][c4 + 0] = vv.x; tile[r][c4 + 1] = vv.y;
    tile[r][c4 + 2] = vv.z; tile[r][c4 + 3] = vv.w;
    __syncthreads();

    uint2 o;
    o.x = pack2(tile[c4 + 0][r], tile[c4 + 1][r]);
    o.y = pack2(tile[c4 + 2][r], tile[c4 + 3][r]);
    *(uint2*)&Wt[lofs + (size_t)(n0 + r) * K + k0 + c4] = o;
}

// ---------------------------------------------------------------------------
// Embedding + positional encoding: h = emb[x] + pe ; also f16 shadow
// ---------------------------------------------------------------------------
__global__ __launch_bounds__(256) void embed_kernel(const int* __restrict__ x,
                                                    const float* __restrict__ emb,
                                                    const float* __restrict__ pe,
                                                    float* __restrict__ h,
                                                    _Float16* __restrict__ h16) {
    int row = blockIdx.x;
    int s   = row & (S_DIM - 1);
    int tok = x[row];
    int d   = threadIdx.x * 4;
    f32x4 e = *(const f32x4*)&emb[(size_t)tok * D_DIM + d];
    f32x4 p = *(const f32x4*)&pe[(size_t)s * D_DIM + d];
    f32x4 o = e + p;
    *(f32x4*)&h[(size_t)row * D_DIM + d] = o;
    uint2 hp; hp.x = pack2(o.x, o.y); hp.y = pack2(o.z, o.w);
    *(uint2*)&h16[(size_t)row * D_DIM + d] = hp;
}

// ---------------------------------------------------------------------------
// GEMM: C[M,N] = epi(A16[M,K] * Wt16[N,K]^T + bias [+ resid]).
// 256 threads = 8 waves; block tile 128(M) x 64(N); K-step 32.
// Both tiles: async DMA f16 global -> LDS, double buffered (ASYNCcnt);
// hot loop is pure DMA-issue + ds_load_b128 fragments + v_wmma.
// EPI: 0 = none, 1 = relu, 2 = add residual.  C / C16 null-checked.
// ---------------------------------------------------------------------------
template <int EPI>
__global__ __launch_bounds__(256) void gemm_kernel(const _Float16* __restrict__ A,
                                                   const _Float16* __restrict__ Wt,
                                                   const float* __restrict__ bias,
                                                   float* __restrict__ C,
                                                   _Float16* __restrict__ C16,
                                                   const float* __restrict__ resid,
                                                   int M, int N, int K) {
    __shared__ __align__(16) _Float16 As[2][128][48];  // [m][k], 96B stride
    __shared__ __align__(16) _Float16 Bt[2][64][48];   // [n][k], 96B stride

    const int tid  = threadIdx.x;
    const int w    = tid >> 5;
    const int lane = tid & 31;
    const int half = lane >> 4;
    const int l16  = lane & 15;
    const int m0   = blockIdx.y * 128;
    const int n0   = blockIdx.x * 64;

    // A staging: 128 rows x 64B = 512 x 16B chunks, 2 per thread
    const int arow = tid >> 2;          // 0..63 (+64 for second chunk)
    const int asub = tid & 3;
    // B staging: 64 rows x 64B = 256 x 16B chunks, 1 per thread
    const int brow = tid >> 2;          // 0..63
    const int bsub = tid & 3;

    // ---- prologue: issue async tiles for k0 = 0 into buffer 0
    #pragma unroll
    for (int j = 0; j < 2; ++j) {
        int row = arow + j * 64;
        unsigned gofs = (unsigned)((((size_t)(m0 + row)) * K + asub * 8) * sizeof(_Float16));
        unsigned lofs = (unsigned)(uintptr_t)&As[0][row][asub * 8];
        asm volatile("global_load_async_to_lds_b128 %0, %1, %2"
                     :: "v"(lofs), "v"(gofs), "s"(A) : "memory");
    }
    {
        unsigned gofs = (unsigned)((((size_t)(n0 + brow)) * K + bsub * 8) * sizeof(_Float16));
        unsigned lofs = (unsigned)(uintptr_t)&Bt[0][brow][bsub * 8];
        asm volatile("global_load_async_to_lds_b128 %0, %1, %2"
                     :: "v"(lofs), "v"(gofs), "s"(Wt) : "memory");
    }

    v8f acc[4] = {};
    int buf = 0;

    for (int k0 = 0; k0 < K; k0 += 32) {
        // tiles for this step must have landed in LDS (per-wave wait + barrier)
        asm volatile("s_wait_asynccnt 0" ::: "memory");
        __syncthreads();

        const int kn = k0 + 32;
        if (kn < K) {
            // issue next tiles into the other buffer (overlaps WMMAs below)
            #pragma unroll
            for (int j = 0; j < 2; ++j) {
                int row = arow + j * 64;
                unsigned gofs = (unsigned)((((size_t)(m0 + row)) * K + kn + asub * 8) * sizeof(_Float16));
                unsigned lofs = (unsigned)(uintptr_t)&As[buf ^ 1][row][asub * 8];
                asm volatile("global_load_async_to_lds_b128 %0, %1, %2"
                             :: "v"(lofs), "v"(gofs), "s"(A) : "memory");
            }
            {
                unsigned gofs = (unsigned)((((size_t)(n0 + brow)) * K + kn + bsub * 8) * sizeof(_Float16));
                unsigned lofs = (unsigned)(uintptr_t)&Bt[buf ^ 1][brow][bsub * 8];
                asm volatile("global_load_async_to_lds_b128 %0, %1, %2"
                             :: "v"(lofs), "v"(gofs), "s"(Wt) : "memory");
            }
            __builtin_prefetch(A  + (size_t)(m0 + arow) * K + kn + 32, 0, 1);
            __builtin_prefetch(Wt + (size_t)(n0 + brow) * K + kn + 32, 0, 1);
        }

        F16x16 af;
        #pragma unroll
        for (int v = 0; v < 8; ++v)
            af.u[v] = *(const unsigned int*)&As[buf][w * 16 + l16][kmap(v, half)];

        #pragma unroll
        for (int nt = 0; nt < 4; ++nt) {
            F16x16 bf;
            #pragma unroll
            for (int v = 0; v < 8; ++v)
                bf.u[v] = *(const unsigned int*)&Bt[buf][nt * 16 + l16][kmap(v, half)];
            acc[nt] = __builtin_amdgcn_wmma_f32_16x16x32_f16(
                false, af.v, false, bf.v, (short)0, acc[nt], false, false);
        }
        __syncthreads();
        buf ^= 1;
    }

    // Epilogue: C-layout lane (m = r + half*8, n = l16) per 16x16 tile
    #pragma unroll
    for (int nt = 0; nt < 4; ++nt) {
        int n = n0 + nt * 16 + l16;
        float bv = bias[n];
        #pragma unroll
        for (int r = 0; r < 8; ++r) {
            int m = m0 + w * 16 + r + half * 8;
            float val = acc[nt][r] + bv;
            if (EPI == 1) val = fmaxf(val, 0.0f);
            if (EPI == 2) val += resid[(size_t)m * N + n];
            if (C)   C[(size_t)m * N + n]   = val;
            if (C16) C16[(size_t)m * N + n] = (_Float16)val;
        }
    }
}

// ---------------------------------------------------------------------------
// Flash-style attention on f16 shadows: one wave per (batch, head, 16-q tile).
// ---------------------------------------------------------------------------
__global__ __launch_bounds__(32) void attn_kernel(const _Float16* __restrict__ Q,
                                                  const _Float16* __restrict__ Kb,
                                                  const _Float16* __restrict__ Vb,
                                                  const int* __restrict__ mask,
                                                  _Float16* __restrict__ O16) {
    __shared__ __align__(16) _Float16 Vt[64][34];   // [dim][key] transposed V chunk
    __shared__ __align__(16) _Float16 Ps[16][34];   // [m][key]   probabilities

    const int lane = threadIdx.x;
    const int half = lane >> 4;
    const int l16  = lane & 15;
    const int qt    = blockIdx.x;
    const int head  = blockIdx.y;
    const int batch = blockIdx.z;
    const size_t base = (size_t)batch * S_DIM;
    const int hoff = head * 64;
    const int qrow = qt * 16;

    // Q fragments: u[0..3] / u[4..7] are two contiguous 16B chunks
    F16x16 qf[2];
    {
        const _Float16* qp = &Q[(base + qrow + l16) * D_DIM + hoff];
        #pragma unroll
        for (int f = 0; f < 2; ++f) {
            uint4 lo = *(const uint4*)(qp + f * 32 + half * 8);
            uint4 hi = *(const uint4*)(qp + f * 32 + 16 + half * 8);
            qf[f].u[0] = lo.x; qf[f].u[1] = lo.y; qf[f].u[2] = lo.z; qf[f].u[3] = lo.w;
            qf[f].u[4] = hi.x; qf[f].u[5] = hi.y; qf[f].u[6] = hi.z; qf[f].u[7] = hi.w;
        }
    }

    float row_max[8], row_sum[8];
    #pragma unroll
    for (int r = 0; r < 8; ++r) { row_max[r] = -1e30f; row_sum[r] = 0.0f; }
    v8f ctx[4] = {};

    const float scale = 0.125f;  // 1/sqrt(64)

    for (int c0idx = 0; c0idx < S_DIM; c0idx += 32) {
        // ---- scores: 16 queries x 32 keys via 4 WMMAs ----
        v8f sc[2] = {};
        #pragma unroll
        for (int t = 0; t < 2; ++t) {
            const _Float16* kp = &Kb[(base + c0idx + t * 16 + l16) * D_DIM + hoff];
            #pragma unroll
            for (int f = 0; f < 2; ++f) {
                uint4 lo = *(const uint4*)(kp + f * 32 + half * 8);
                uint4 hi = *(const uint4*)(kp + f * 32 + 16 + half * 8);
                F16x16 kf;
                kf.u[0] = lo.x; kf.u[1] = lo.y; kf.u[2] = lo.z; kf.u[3] = lo.w;
                kf.u[4] = hi.x; kf.u[5] = hi.y; kf.u[6] = hi.z; kf.u[7] = hi.w;
                sc[t] = __builtin_amdgcn_wmma_f32_16x16x32_f16(
                    false, qf[f].v, false, kf.v, (short)0, sc[t], false, false);
            }
        }

        int mk0 = mask[batch * S_DIM + c0idx + l16];
        int mk1 = mask[batch * S_DIM + c0idx + 16 + l16];

        // ---- online softmax (rows r+half*8 live across the 16-lane half) ----
        float nm[8];
        #pragma unroll
        for (int r = 0; r < 8; ++r) {
            float s0 = sc[0][r] * scale; if (mk0 == 0) s0 = -1e9f;
            float s1 = sc[1][r] * scale; if (mk1 == 0) s1 = -1e9f;
            sc[0][r] = s0; sc[1][r] = s1;
            float t = fmaxf(s0, s1);
            t = fmaxf(t, __shfl_xor(t, 1, 32));
            t = fmaxf(t, __shfl_xor(t, 2, 32));
            t = fmaxf(t, __shfl_xor(t, 4, 32));
            t = fmaxf(t, __shfl_xor(t, 8, 32));
            nm[r] = fmaxf(row_max[r], t);
        }
        #pragma unroll
        for (int r = 0; r < 8; ++r) {
            float alpha = __expf(row_max[r] - nm[r]);
            row_max[r] = nm[r];
            float p0 = __expf(sc[0][r] - nm[r]);
            float p1 = __expf(sc[1][r] - nm[r]);
            sc[0][r] = p0; sc[1][r] = p1;
            float ts = p0 + p1;
            ts += __shfl_xor(ts, 1, 32);
            ts += __shfl_xor(ts, 2, 32);
            ts += __shfl_xor(ts, 4, 32);
            ts += __shfl_xor(ts, 8, 32);
            row_sum[r] = row_sum[r] * alpha + ts;
            #pragma unroll
            for (int f = 0; f < 4; ++f) ctx[f][r] *= alpha;
        }

        // ---- stage P (C-layout -> LDS row major) and V (transposed) ----
        #pragma unroll
        for (int r = 0; r < 8; ++r) {
            Ps[r + half * 8][l16]      = (_Float16)sc[0][r];
            Ps[r + half * 8][16 + l16] = (_Float16)sc[1][r];
        }
        {
            const _Float16* vrow = &Vb[(base + c0idx + lane) * D_DIM + hoff];
            #pragma unroll
            for (int c = 0; c < 8; ++c) {
                uint4 q4 = *(const uint4*)(vrow + c * 8);
                unsigned uu[4] = {q4.x, q4.y, q4.z, q4.w};
                #pragma unroll
                for (int i = 0; i < 4; ++i) {
                    Pack2 pp; pp.u = uu[i];
                    Vt[c * 8 + i * 2 + 0][lane] = pp.h[0];
                    Vt[c * 8 + i * 2 + 1][lane] = pp.h[1];
                }
            }
        }
        __syncthreads();

        // ---- ctx += P(16x32) * V(32x64) via 4 WMMAs ----
        F16x16 pf;
        #pragma unroll
        for (int v = 0; v < 8; ++v)
            pf.u[v] = *(const unsigned int*)&Ps[l16][kmap(v, half)];
        #pragma unroll
        for (int f = 0; f < 4; ++f) {
            F16x16 vf;
            #pragma unroll
            for (int v = 0; v < 8; ++v)
                vf.u[v] = *(const unsigned int*)&Vt[f * 16 + l16][kmap(v, half)];
            ctx[f] = __builtin_amdgcn_wmma_f32_16x16x32_f16(
                false, pf.v, false, vf.v, (short)0, ctx[f], false, false);
        }
        __syncthreads();
    }

    // ---- normalize and write ctx (f16 shadow feeds the Wo GEMM) ----
    #pragma unroll
    for (int r = 0; r < 8; ++r) {
        float inv = 1.0f / row_sum[r];
        #pragma unroll
        for (int f = 0; f < 4; ++f) {
            size_t idx = (base + qrow + r + half * 8) * D_DIM + hoff + f * 16 + l16;
            O16[idx] = (_Float16)(ctx[f][r] * inv);
        }
    }
}

// ---------------------------------------------------------------------------
// LayerNorm over last dim (1024): one 256-thread block per row.
// ---------------------------------------------------------------------------
__global__ __launch_bounds__(256) void ln_kernel(const float* __restrict__ X,
                                                 const float* __restrict__ g,
                                                 const float* __restrict__ b,
                                                 float* __restrict__ O,
                                                 _Float16* __restrict__ O16) {
    __shared__ float red[256];
    const int tid = threadIdx.x;
    const int row = blockIdx.x;
    const int d0  = tid * 4;

    f32x4 v = *(const f32x4*)(X + (size_t)row * D_DIM + d0);
    float s = v.x + v.y + v.z + v.w;
    red[tid] = s;
    __syncthreads();
    for (int off = 128; off > 0; off >>= 1) {
        if (tid < off) red[tid] += red[tid + off];
        __syncthreads();
    }
    float mu = red[0] * (1.0f / D_DIM);
    __syncthreads();

    f32x4 dv = v - mu;
    red[tid] = dv.x * dv.x + dv.y * dv.y + dv.z * dv.z + dv.w * dv.w;
    __syncthreads();
    for (int off = 128; off > 0; off >>= 1) {
        if (tid < off) red[tid] += red[tid + off];
        __syncthreads();
    }
    float var = red[0] * (1.0f / D_DIM);
    float inv = rsqrtf(var + 1e-5f);

    f32x4 gg = *(const f32x4*)(g + d0);
    f32x4 bb = *(const f32x4*)(b + d0);
    f32x4 o  = dv * inv * gg + bb;
    *(f32x4*)(O + (size_t)row * D_DIM + d0) = o;
    if (O16) {
        uint2 hp; hp.x = pack2(o.x, o.y); hp.y = pack2(o.z, o.w);
        *(uint2*)&O16[(size_t)row * D_DIM + d0] = hp;
    }
}

// ---------------------------------------------------------------------------
// Host orchestration
// ---------------------------------------------------------------------------
extern "C" void kernel_launch(void* const* d_in, const int* in_sizes, int n_in,
                              void* d_out, int out_size, void* d_ws, size_t ws_size,
                              hipStream_t stream) {
    const int*   x     = (const int*)d_in[0];
    const int*   mask  = (const int*)d_in[1];
    const float* pe    = (const float*)d_in[2];
    const float* emb   = (const float*)d_in[3];
    const float* Wq    = (const float*)d_in[4];
    const float* bq    = (const float*)d_in[5];
    const float* Wk    = (const float*)d_in[6];
    const float* bk    = (const float*)d_in[7];
    const float* Wv    = (const float*)d_in[8];
    const float* bv    = (const float*)d_in[9];
    const float* Wo    = (const float*)d_in[10];
    const float* bo    = (const float*)d_in[11];
    const float* W1    = (const float*)d_in[12];
    const float* b1    = (const float*)d_in[13];
    const float* W2    = (const float*)d_in[14];
    const float* b2    = (const float*)d_in[15];
    const float* g1    = (const float*)d_in[16];
    const float* be1   = (const float*)d_in[17];
    const float* g2    = (const float*)d_in[18];
    const float* be2   = (const float*)d_in[19];

    float* ws  = (float*)d_ws;
    float* h   = ws;                        // NT * D fp32 (residual source)
    float* t2  = h + (size_t)NT * D_DIM;    // NT * D fp32 (pre-LN sum)
    // f16 shadow activations
    _Float16* h16  = (_Float16*)(t2 + (size_t)NT * D_DIM);
    _Float16* q16  = h16 + (size_t)NT * D_DIM;
    _Float16* k16  = q16 + (size_t)NT * D_DIM;
    _Float16* v16  = k16 + (size_t)NT * D_DIM;
    _Float16* c16  = v16 + (size_t)NT * D_DIM;   // attention output
    _Float16* t116 = c16 + (size_t)NT * D_DIM;   // NT * F
    // f16 pre-transposed weights [N][K]
    _Float16* wqt = t116 + (size_t)NT * F_DIM;         // L*D*D
    _Float16* wkt = wqt + (size_t)L_DIM * D_DIM * D_DIM;
    _Float16* wvt = wkt + (size_t)L_DIM * D_DIM * D_DIM;
    _Float16* wot = wvt + (size_t)L_DIM * D_DIM * D_DIM;
    _Float16* w1t = wot + (size_t)L_DIM * D_DIM * D_DIM;  // [F][D] per layer
    _Float16* w2t = w1t + (size_t)L_DIM * D_DIM * F_DIM;  // [D][F] per layer

    // ---- one-time weight convert+transpose (amortized over 64 M-tiles) ----
    dim3 gWdd(D_DIM / 32, D_DIM / 32, L_DIM);
    dim3 gWdf(D_DIM / 32, F_DIM / 32, L_DIM);   // W1: K=D, N=F
    dim3 gWfd(F_DIM / 32, D_DIM / 32, L_DIM);   // W2: K=F, N=D
    wconv_kernel<<<gWdd, 256, 0, stream>>>(Wq, wqt, D_DIM, D_DIM);
    wconv_kernel<<<gWdd, 256, 0, stream>>>(Wk, wkt, D_DIM, D_DIM);
    wconv_kernel<<<gWdd, 256, 0, stream>>>(Wv, wvt, D_DIM, D_DIM);
    wconv_kernel<<<gWdd, 256, 0, stream>>>(Wo, wot, D_DIM, D_DIM);
    wconv_kernel<<<gWdf, 256, 0, stream>>>(W1, w1t, D_DIM, F_DIM);
    wconv_kernel<<<gWfd, 256, 0, stream>>>(W2, w2t, F_DIM, D_DIM);

    embed_kernel<<<NT, 256, 0, stream>>>(x, emb, pe, h, h16);

    dim3 gD(D_DIM / 64, NT / 128);   // GEMMs with N = 1024
    dim3 gF(F_DIM / 64, NT / 128);   // GEMM  with N = 4096
    dim3 gA(S_DIM / 16, H_DIM, B_DIM);

    for (int l = 0; l < L_DIM; ++l) {
        const _Float16* wqt_l = wqt + (size_t)l * D_DIM * D_DIM;
        const _Float16* wkt_l = wkt + (size_t)l * D_DIM * D_DIM;
        const _Float16* wvt_l = wvt + (size_t)l * D_DIM * D_DIM;
        const _Float16* wot_l = wot + (size_t)l * D_DIM * D_DIM;
        const _Float16* w1t_l = w1t + (size_t)l * D_DIM * F_DIM;
        const _Float16* w2t_l = w2t + (size_t)l * F_DIM * D_DIM;
        const float* bq_l = bq + (size_t)l * D_DIM;
        const float* bk_l = bk + (size_t)l * D_DIM;
        const float* bv_l = bv + (size_t)l * D_DIM;
        const float* bo_l = bo + (size_t)l * D_DIM;
        const float* b1_l = b1 + (size_t)l * F_DIM;
        const float* b2_l = b2 + (size_t)l * D_DIM;

        gemm_kernel<0><<<gD, 256, 0, stream>>>(h16, wqt_l, bq_l, nullptr, q16, nullptr, NT, D_DIM, D_DIM);
        gemm_kernel<0><<<gD, 256, 0, stream>>>(h16, wkt_l, bk_l, nullptr, k16, nullptr, NT, D_DIM, D_DIM);
        gemm_kernel<0><<<gD, 256, 0, stream>>>(h16, wvt_l, bv_l, nullptr, v16, nullptr, NT, D_DIM, D_DIM);

        attn_kernel<<<gA, 32, 0, stream>>>(q16, k16, v16, mask, c16);

        // t2 = ctx @ Wo + bo + h ; h = LN(t2)
        gemm_kernel<2><<<gD, 256, 0, stream>>>(c16, wot_l, bo_l, t2, nullptr, h, NT, D_DIM, D_DIM);
        ln_kernel<<<NT, 256, 0, stream>>>(t2, g1 + (size_t)l * D_DIM, be1 + (size_t)l * D_DIM, h, h16);

        // t1 = relu(h @ W1 + b1) (f16 only) ; t2 = t1 @ W2 + b2 + h ; h = LN(t2)
        gemm_kernel<1><<<gF, 256, 0, stream>>>(h16, w1t_l, b1_l, nullptr, t116, nullptr, NT, F_DIM, D_DIM);
        gemm_kernel<2><<<gD, 256, 0, stream>>>(t116, w2t_l, b2_l, t2, nullptr, h, NT, D_DIM, F_DIM);

        float* ln_out = (l == L_DIM - 1) ? (float*)d_out : h;
        _Float16* ln_out16 = (l == L_DIM - 1) ? nullptr : h16;
        ln_kernel<<<NT, 256, 0, stream>>>(t2, g2 + (size_t)l * D_DIM, be2 + (size_t)l * D_DIM, ln_out, ln_out16);
    }
}
